// GlobalConv1d_7224134992309
// MI455X (gfx1250) — matrix operand order repaired and checked by reading, not compiled, not run on hardware
//
#include <hip/hip_runtime.h>

// Problem constants (B, C, OC, D) = (16, 16, 16, 1024)
constexpr int kB  = 16;
constexpr int kC  = 16;
constexpr int kOC = 16;
constexpr int kD  = 1024;
constexpr int kDP = 1088;            // padded w row: D + 64 wrap-replicated elems
constexpr int kHalfD = kD / 2;       // 512 output shifts
constexpr int kLdsRowX = 1032;       // 1024 + 8-half pad (bank-conflict relief)
constexpr int kLdsRowW = 1096;       // 1088 + 8-half pad

typedef __attribute__((ext_vector_type(16))) __bf16       v16bf;
typedef __attribute__((ext_vector_type(8)))  float        v8f;
typedef __attribute__((ext_vector_type(4)))  unsigned int v4u;
typedef __attribute__((ext_vector_type(8)))  unsigned int v8u;
typedef __attribute__((ext_vector_type(4)))  int          v4i;

typedef __attribute__((address_space(1))) v4i* gvec4_ptr;  // global int4*
typedef __attribute__((address_space(3))) v4i* lvec4_ptr;  // LDS int4*

#if defined(__gfx1250__) && __has_builtin(__builtin_amdgcn_global_load_async_to_lds_b128) && \
    __has_builtin(__builtin_amdgcn_s_wait_asynccnt)
#define USE_ASYNC_LDS 1
#else
#define USE_ASYNC_LDS 0
#endif

__device__ __forceinline__ unsigned int bf16rne(float f) {
  unsigned int u = __float_as_uint(f);
  return (u + 0x7FFFu + ((u >> 16) & 1u)) >> 16;   // round-to-nearest-even
}
__device__ __forceinline__ unsigned int pk2(float a, float b) {
  return bf16rne(a) | (bf16rne(b) << 16);
}

// ---- one-time f32 -> bf16 conversion of x (262144 elems) ----
__global__ __launch_bounds__(256) void cvt_x_kernel(const float* __restrict__ x,
                                                    unsigned short* __restrict__ xbf) {
  int t = blockIdx.x * 256 + threadIdx.x;       // 65536 threads * 4 floats
  const float4 v = ((const float4*)x)[t];
  uint2 p;
  p.x = pk2(v.x, v.y);
  p.y = pk2(v.z, v.w);
  ((uint2*)xbf)[t] = p;
}

// ---- f32 -> bf16 + circular pad of w: row stride 1088, tail = wrap copy ----
__global__ __launch_bounds__(256) void cvt_w_kernel(const float* __restrict__ w,
                                                    unsigned int* __restrict__ wbf) {
  int row = blockIdx.x;                          // 256 rows = (o*16 + c)
  const float* src = w + row * kD;
  unsigned int* dst = wbf + row * (kDP / 2);
  for (int dj = threadIdx.x; dj < kDP / 2; dj += 256) {
    int e = (dj * 2) & (kD - 1);                 // pairs never straddle the wrap
    dst[dj] = pk2(src[e], src[e + 1]);
  }
}

// 16-byte global -> LDS copy: async (CDNA5 LDS-direct DMA) or sync fallback.
__device__ __forceinline__ void stage16(const unsigned short* g, unsigned short* l) {
#if USE_ASYNC_LDS
  __builtin_amdgcn_global_load_async_to_lds_b128(
      (gvec4_ptr)g, (lvec4_ptr)l, 0, 0);
#else
  *(v4u*)l = *(const v4u*)g;
#endif
}

// Stage x[:, c, :] (16x1024) and padded w[:, c, :] (16x1088) slabs into LDS.
__device__ __forceinline__ void stage_slab(const unsigned short* __restrict__ xbf,
                                           const unsigned short* __restrict__ wbf,
                                           int c, int tid,
                                           unsigned short* lx, unsigned short* lw) {
  #pragma unroll
  for (int it = 0; it < 8; ++it) {               // 2048 b128 chunks of x slab
    int q = it * 256 + tid;
    int row = q >> 7, col = q & 127;
    stage16(xbf + row * (kC * kD) + c * kD + col * 8, lx + row * kLdsRowX + col * 8);
  }
  for (int q = tid; q < 2176; q += 256) {        // 16 rows x 136 b128 chunks of w slab
    int row = q / 136, col = q - row * 136;
    stage16(wbf + (row * kC + c) * kDP + col * 8, lw + row * kLdsRowW + col * 8);
  }
}

__device__ __forceinline__ void wait_stage() {
#if USE_ASYNC_LDS
  __builtin_amdgcn_s_wait_asynccnt(0);
#endif
}

// One 16x16x32 bf16 WMMA: A from LDS x-slab, B from LDS w-slab.
__device__ __forceinline__ v8f wmma_bf16(const unsigned short* ap, const unsigned int* bp,
                                         v8f acc) {
  v4u a0 = *(const v4u*)ap;                       // K [koff .. koff+7]
  v4u a1 = *(const v4u*)(ap + 8);                 // K [koff+8 .. koff+15]
  v8u araw = __builtin_shufflevector(a0, a1, 0, 1, 2, 3, 4, 5, 6, 7);
  v8u braw;
  #pragma unroll
  for (int q = 0; q < 8; ++q) braw[q] = bp[q];    // dword loads: aligned for any shift
  return __builtin_amdgcn_wmma_f32_16x16x32_bf16(
      false, __builtin_bit_cast(v16bf, araw),
      false, __builtin_bit_cast(v16bf, braw),
      (short)0, acc, false, false);
}

// ---- main: 512 shifts, one 16x16xK=16384 GEMM per wave via bf16 WMMA ----
__global__ __launch_bounds__(256) void conv_wmma_kernel(
    const unsigned short* __restrict__ xbf,
    const unsigned short* __restrict__ wbf,
    float* __restrict__ out) {
  __shared__ unsigned short lds_x[2][16 * kLdsRowX];   // 2 x 33 KB
  __shared__ unsigned short lds_w[2][16 * kLdsRowW];   // 2 x 35 KB

  const int tid  = threadIdx.x;
  const int lane = tid & 31;
  const int wave = tid >> 5;
  const int i    = blockIdx.x * 8 + wave;          // shift index 0..511
  const int m    = lane & 15;                      // A row (b) / B col (o)
  const int hi   = lane >> 4;                      // lane half selects K sub-block
  const int s    = __builtin_amdgcn_readfirstlane(2 * i);  // scalar shift
  const int kcw  = (kD - s + 31) >> 5;             // first k-chunk that wraps

  v8f acc = {};

  // Prologue: stage channel 0 into buffer 0.
  stage_slab(xbf, wbf, 0, tid, lds_x[0], lds_w[0]);
  wait_stage();
  __syncthreads();

  for (int c = 0; c < kC; ++c) {
    const int cur = c & 1;
    // Overlap: issue async staging of channel c+1 into the other buffer.
    if (c + 1 < kC)
      stage_slab(xbf, wbf, c + 1, tid, lds_x[cur ^ 1], lds_w[cur ^ 1]);

    const unsigned short* ax = &lds_x[cur][m * kLdsRowX + hi * 16];
    const unsigned int*   bw = (const unsigned int*)&lds_w[cur][0]
                               + m * (kLdsRowW / 2) + hi * 8;
    // Segment 1: base = s + 32*kc (no wrap).  Segment 2: base - D.
    const int dw1 = s >> 1;
    const int dw2 = (s - kD) >> 1;
    #pragma unroll 4
    for (int kc = 0; kc < kcw; ++kc)
      acc = wmma_bf16(ax + 32 * kc, bw + (dw1 + 16 * kc), acc);
    #pragma unroll 4
    for (int kc = kcw; kc < 32; ++kc)
      acc = wmma_bf16(ax + 32 * kc, bw + (dw2 + 16 * kc), acc);

    wait_stage();        // c+1 slab resident before anyone reads it
    __syncthreads();     // and everyone done reading slab c before reuse
  }

  // D layout: lane holds col n = m; VGPR r holds row (r + 8*hi). Apply 1/sqrt(C).
  float* op = out + (size_t)(hi * 8) * (kOC * kHalfD) + m * kHalfD + i;
  #pragma unroll
  for (int r = 0; r < 8; ++r)
    op[(size_t)r * (kOC * kHalfD)] = acc[r] * 0.25f;
}

extern "C" void kernel_launch(void* const* d_in, const int* in_sizes, int n_in,
                              void* d_out, int out_size, void* d_ws, size_t ws_size,
                              hipStream_t stream) {
  const float* x = (const float*)d_in[0];   // (16, 16, 1024) f32
  const float* w = (const float*)d_in[1];   // (16, 16, 1, 1024) f32
  float* out = (float*)d_out;               // (16, 16, 512) f32

  unsigned short* xbf = (unsigned short*)d_ws;          // 524288 B
  unsigned short* wbf = xbf + (kB * kC * kD);           // 557056 B padded w
  (void)in_sizes; (void)n_in; (void)out_size; (void)ws_size;

  cvt_x_kernel<<<256, 256, 0, stream>>>(x, xbf);
  cvt_w_kernel<<<256, 256, 0, stream>>>(w, (unsigned int*)wbf);
  conv_wmma_kernel<<<64, 256, 0, stream>>>(xbf, wbf, out);
}
